// LAS_10771777978982
// MI455X (gfx1250) — matrix operand order repaired and checked by reading, not compiled, not run on hardware
//
#include <hip/hip_runtime.h>
#include <stdint.h>
#include <stddef.h>

// Problem dims (from reference)
#define B_     32
#define T_     256
#define S_     1600
#define DENC_  1024
#define V_     1000
#define EMB_   512
#define PRE_   256
#define LSTMU_ 1024
#define A_     128
#define F_     32
#define KCONV_ 31
#define GATES_ 4096   // 4*LSTMU
#define SCHNK_ 25     // S_ / 64 chunks for ctx partial reduction

typedef __attribute__((ext_vector_type(16))) __bf16 v16bf;
typedef __attribute__((ext_vector_type(8)))  __bf16 v8bf;
typedef __attribute__((ext_vector_type(8)))  float  v8f;

// ---------- bf16 helpers (manual RNE, storage = uint16_t) ----------
__device__ __forceinline__ uint16_t f2bf(float f) {
  union { float f; uint32_t u; } x; x.f = f;
  uint32_t u = x.u;
  uint32_t r = (u + 0x7fffu + ((u >> 16) & 1u)) >> 16;
  return (uint16_t)r;
}
__device__ __forceinline__ float bf2f(uint16_t h) {
  union { uint32_t u; float f; } x; x.u = ((uint32_t)h) << 16; return x.f;
}

// ---------- generic WMMA GEMM: Z[r0..r0+31, nt*16..+15] = X(bf16) @ Wpk(bf16) + bias
// One wave (32 threads) per 32x16 tile; two v_wmma_f32_16x16x32_bf16 per K-tile.
// Wpk is pre-packed so each lane's B-fragment (16 bf16) is contiguous:
//   offset = ((kt*ntiles + nt)*32 + lane)*16 ;  lane holds n = nt*16 + (lane&15),
//   k = kt*32 + (e&8 ? 16 : 0) + (lane<16 ? 0 : 8) + (e&7)  (e = 0..15)
// A fragment (16-bit A 16x32 layout): lane m = lane&15, base=(lane<16)?0:8,
//   elems 0..7 <- k0+base..+7, elems 8..15 <- k0+16+base..+7 (two 16B loads).
// store_bf16: write D as bf16 (used for the keys GEMM to keep keys L2-resident).
__global__ __launch_bounds__(32)
void wmma_gemm32(const uint16_t* __restrict__ Xu, int ldx,
                 const uint16_t* __restrict__ Wu, int ntiles,
                 const float* __restrict__ bias,
                 float* __restrict__ Z, long long ldz,
                 int K, int n_limit, int store_bf16)
{
  const __bf16* X = (const __bf16*)Xu;
  const __bf16* W = (const __bf16*)Wu;
  int lane = threadIdx.x;
  int nt   = blockIdx.x;
  int r0   = blockIdx.y * 32;
  int ncol = nt * 16 + (lane & 15);
  float bval = 0.f;
  if (bias && ncol < n_limit) bval = bias[ncol];
  v8f acc0, acc1;
#pragma unroll
  for (int i = 0; i < 8; ++i) { acc0[i] = bval; acc1[i] = bval; }

  int m0   = r0 + (lane & 15);
  int aoff = (lane < 16) ? 0 : 8;
  const __bf16* xa0 = X + (size_t)m0 * ldx + aoff;
  const __bf16* xa1 = xa0 + (size_t)16 * ldx;
  const __bf16* wp  = W + (size_t)nt * 512 + (size_t)lane * 16;
  size_t wstride = (size_t)ntiles * 512;

  int ktiles = K >> 5;
  for (int kt = 0; kt < ktiles; ++kt) {
    int k0 = kt << 5;
    v8bf a0lo = *(const v8bf*)(xa0 + k0);
    v8bf a0hi = *(const v8bf*)(xa0 + k0 + 16);
    v8bf a1lo = *(const v8bf*)(xa1 + k0);
    v8bf a1hi = *(const v8bf*)(xa1 + k0 + 16);
    v16bf a0 = __builtin_shufflevector(a0lo, a0hi, 0,1,2,3,4,5,6,7,8,9,10,11,12,13,14,15);
    v16bf a1 = __builtin_shufflevector(a1lo, a1hi, 0,1,2,3,4,5,6,7,8,9,10,11,12,13,14,15);
    v16bf bf = *(const v16bf*)(wp);
    wp += wstride;
    acc0 = __builtin_amdgcn_wmma_f32_16x16x32_bf16(false, a0, false, bf, (short)0, acc0, false, false);
    acc1 = __builtin_amdgcn_wmma_f32_16x16x32_bf16(false, a1, false, bf, (short)0, acc1, false, false);
  }

  if (ncol < n_limit) {
    int rb = (lane < 16) ? 0 : 8;   // C/D layout: vgpr v -> row v (+8 for hi lanes)
    if (store_bf16) {
      uint16_t* Zb = (uint16_t*)Z;
#pragma unroll
      for (int v = 0; v < 8; ++v)
        Zb[(size_t)(r0 + rb + v) * ldz + ncol] = f2bf(acc0[v]);
#pragma unroll
      for (int v = 0; v < 8; ++v)
        Zb[(size_t)(r0 + 16 + rb + v) * ldz + ncol] = f2bf(acc1[v]);
    } else {
#pragma unroll
      for (int v = 0; v < 8; ++v)
        Z[(size_t)(r0 + rb + v) * ldz + ncol] = acc0[v];
#pragma unroll
      for (int v = 0; v < 8; ++v)
        Z[(size_t)(r0 + 16 + rb + v) * ldz + ncol] = acc1[v];
    }
  }
}

// ---------- weight packing into WMMA B-fragment order, fp32 -> bf16 ----------
// Supports stacked [S0;S1] (for W;U concat) and N padding (for Wc: 1000 -> 1008).
__global__ void pack_w(const float* __restrict__ s0, int rows0,
                       const float* __restrict__ s1,
                       int Nsrc, int Ndst, uint16_t* __restrict__ dst, long long total)
{
  long long idx = (long long)blockIdx.x * blockDim.x + threadIdx.x;
  long long stride = (long long)gridDim.x * blockDim.x;
  int ntiles = Ndst >> 4;
  for (; idx < total; idx += stride) {
    int e    = (int)(idx & 15);
    int lane = (int)((idx >> 4) & 31);
    long long rest = idx >> 9;
    int nt = (int)(rest % ntiles);
    int kt = (int)(rest / ntiles);
    int k = (kt << 5) + ((e & 8) << 1) + ((lane & 16) ? 8 : 0) + (e & 7);
    int n = (nt << 4) + (lane & 15);
    float v = 0.f;
    if (n < Nsrc) v = (k < rows0) ? s0[(size_t)k * Nsrc + n]
                                  : s1[(size_t)(k - rows0) * Nsrc + n];
    dst[idx] = f2bf(v);
  }
}

__global__ void cast_f32_bf16(const float* __restrict__ src, uint16_t* __restrict__ dst, long long n)
{
  long long i = (long long)blockIdx.x * blockDim.x + threadIdx.x;
  long long st = (long long)gridDim.x * blockDim.x;
  for (; i < n; i += st) dst[i] = f2bf(src[i]);
}

__global__ void zero_f32(float* __restrict__ p, long long n)
{
  long long i = (long long)blockIdx.x * blockDim.x + threadIdx.x;
  long long st = (long long)gridDim.x * blockDim.x;
  for (; i < n; i += st) p[i] = 0.f;
}

// copy 32 x cols f32 rows into bf16 xcat slab at column offset
__global__ void cast_rows(const float* __restrict__ src, uint16_t* __restrict__ dst,
                          int dstld, int dstoff, int cols)
{
  int idx = blockIdx.x * blockDim.x + threadIdx.x;
  if (idx >= B_ * cols) return;
  int r = idx / cols, c = idx - r * cols;
  dst[(size_t)r * dstld + dstoff + c] = f2bf(src[(size_t)r * cols + c]);
}

// ---------- prenet: emb lookup -> FC(512->256) relu -> FC(256->256) relu -> bf16 xcat[:,0:256]
__global__ __launch_bounds__(256)
void prenet_kernel(const int* __restrict__ tokens, int t,
                   const float* __restrict__ emb,
                   const float* __restrict__ Wp1, const float* __restrict__ bp1,
                   const float* __restrict__ Wp2, const float* __restrict__ bp2,
                   uint16_t* __restrict__ xcat, int xld)
{
  __shared__ float xs[EMB_];
  __shared__ float p1[PRE_];
  int b = blockIdx.x, tid = threadIdx.x;
  int tok = tokens[b * T_ + t];
  for (int j = tid; j < EMB_; j += 256) xs[j] = emb[(size_t)tok * EMB_ + j];
  __syncthreads();
  float acc = bp1[tid];
  for (int j = 0; j < EMB_; ++j) acc += xs[j] * Wp1[(size_t)j * PRE_ + tid];
  p1[tid] = fmaxf(acc, 0.f);
  __syncthreads();
  float acc2 = bp2[tid];
  for (int j = 0; j < PRE_; ++j) acc2 += p1[j] * Wp2[(size_t)j * PRE_ + tid];
  xcat[(size_t)b * xld + tid] = f2bf(fmaxf(acc2, 0.f));
}

// ---------- LSTM elementwise: z[32,4096] -> h,c [32,1024]
__global__ __launch_bounds__(256)
void lstm_update(const float* __restrict__ z, float* __restrict__ h, float* __restrict__ c)
{
  int idx = blockIdx.x * blockDim.x + threadIdx.x;   // 32*1024 threads
  int b = idx >> 10, u = idx & 1023;
  const float* zr = z + (size_t)b * GATES_;
  float gi = zr[u];
  float gf = zr[LSTMU_ + u];
  float gg = zr[2 * LSTMU_ + u];
  float go = zr[3 * LSTMU_ + u];
  float si = 1.f / (1.f + __expf(-gi));
  float sf = 1.f / (1.f + __expf(-gf));
  float so = 1.f / (1.f + __expf(-go));
  float cn = sf * c[idx] + si * tanhf(gg);
  c[idx] = cn;
  h[idx] = so * tanhf(cn);
}

// ---------- pq = h_a @ Wq  [32,128]
__global__ __launch_bounds__(128)
void pq_kernel(const float* __restrict__ h, const float* __restrict__ Wq,
               float* __restrict__ pq)
{
  int b = blockIdx.x, a = threadIdx.x;
  float acc = 0.f;
  for (int j = 0; j < LSTMU_; ++j) acc += h[(size_t)b * LSTMU_ + j] * Wq[(size_t)j * A_ + a];
  pq[b * A_ + a] = acc;
}

// ---------- energy: conv(ast)->f, e = tanh(keys + pq + f@Wloc) @ Wv + bv, masked
// keys are bf16 (kept L2-resident: 13 MB).
__global__ __launch_bounds__(128)
void energy_kernel(const uint16_t* __restrict__ keys, const float* __restrict__ pq,
                   const float* __restrict__ ast, const float* __restrict__ ck,
                   const float* __restrict__ Wloc, const float* __restrict__ Wv,
                   const float* __restrict__ bv, const int* __restrict__ memlen,
                   float* __restrict__ e)
{
  int idx = blockIdx.x * blockDim.x + threadIdx.x;
  if (idx >= B_ * S_) return;
  int b = idx / S_, s = idx - b * S_;
  if (s >= memlen[b]) { e[idx] = -1e9f; return; }
  float fc[F_];
#pragma unroll
  for (int c = 0; c < F_; ++c) fc[c] = 0.f;
  const float* arow = ast + (size_t)b * S_;
  for (int k = 0; k < KCONV_; ++k) {
    int sp = s + k - (KCONV_ / 2);
    if (sp < 0 || sp >= S_) continue;
    float av = arow[sp];
#pragma unroll
    for (int c = 0; c < F_; ++c) fc[c] += av * ck[k * F_ + c];
  }
  const uint16_t* krow = keys + (size_t)idx * A_;
  const float* pqb  = pq + b * A_;
  float acc = 0.f;
  for (int a = 0; a < A_; ++a) {
    float val = bf2f(krow[a]) + pqb[a];
#pragma unroll
    for (int c = 0; c < F_; ++c) val += fc[c] * Wloc[c * A_ + a];
    acc += tanhf(val) * Wv[a];
  }
  e[idx] = acc + bv[0];
}

// ---------- softmax over S per batch row; also ast += al
__global__ __launch_bounds__(256)
void softmax_kernel(const float* __restrict__ e, float* __restrict__ al,
                    float* __restrict__ ast)
{
  __shared__ float red[256];
  int b = blockIdx.x, tid = threadIdx.x;
  const float* er = e + (size_t)b * S_;
  float m = -3.4e38f;
  for (int s = tid; s < S_; s += 256) m = fmaxf(m, er[s]);
  red[tid] = m; __syncthreads();
  for (int st = 128; st > 0; st >>= 1) { if (tid < st) red[tid] = fmaxf(red[tid], red[tid + st]); __syncthreads(); }
  m = red[0]; __syncthreads();
  float sum = 0.f;
  for (int s = tid; s < S_; s += 256) sum += __expf(er[s] - m);
  red[tid] = sum; __syncthreads();
  for (int st = 128; st > 0; st >>= 1) { if (tid < st) red[tid] += red[tid + st]; __syncthreads(); }
  float inv = 1.f / red[0];
  for (int s = tid; s < S_; s += 256) {
    float a = __expf(er[s] - m) * inv;
    al[(size_t)b * S_ + s] = a;
    ast[(size_t)b * S_ + s] += a;
  }
}

// ---------- ctx stage 1: part[b,sc,d] = sum_{s in chunk} al[b,s]*mem[b,s,d]
// The HBM/L2-bandwidth-critical op: grid (B, 25 chunks of 64 s) x 128 thr,
// each thread owns 8 consecutive d and loads them as one 16-byte uint4
// (512 B per wave-instruction). Deterministic (no float atomics).
__global__ __launch_bounds__(128)
void ctx_partial_kernel(const float* __restrict__ al, const uint16_t* __restrict__ mem,
                        float* __restrict__ part)   // [B][SCHNK_][DENC]
{
  __shared__ float als[64];
  int b = blockIdx.x, sc = blockIdx.y, tid = threadIdx.x;
  int s0 = sc * 64;
  if (tid < 64) als[tid] = al[(size_t)b * S_ + s0 + tid];
  __syncthreads();
  int d0 = tid * 8;
  const uint16_t* mr = mem + ((size_t)b * S_ + s0) * DENC_ + d0;
  float acc[8];
#pragma unroll
  for (int i = 0; i < 8; ++i) acc[i] = 0.f;
  for (int si = 0; si < 64; ++si) {
    float a = als[si];
    uint4 pk = *(const uint4*)(mr + (size_t)si * DENC_);
    const uint16_t* q = (const uint16_t*)&pk;
#pragma unroll
    for (int i = 0; i < 8; ++i) acc[i] += a * bf2f(q[i]);
  }
  float* pr = part + ((size_t)b * SCHNK_ + sc) * DENC_ + d0;
#pragma unroll
  for (int i = 0; i < 8; ++i) pr[i] = acc[i];
}

// ---------- ctx stage 2: deterministic serial sum of the 25 partials
__global__ __launch_bounds__(256)
void ctx_reduce_kernel(const float* __restrict__ part, float* __restrict__ ctx)
{
  int idx = blockIdx.x * blockDim.x + threadIdx.x;  // 32*1024 threads
  int b = idx >> 10, d = idx & 1023;
  const float* pr = part + (size_t)b * SCHNK_ * DENC_ + d;
  float acc = 0.f;
#pragma unroll
  for (int i = 0; i < SCHNK_; ++i) acc += pr[(size_t)i * DENC_];
  ctx[idx] = acc;
}

// ---------- stop token: stp = [h1, cls] @ Ws + bs
__global__ __launch_bounds__(256)
void stp_kernel(const float* __restrict__ h1, const float* __restrict__ cls_base,
                long long cls_ld, const float* __restrict__ Ws, const float* __restrict__ bs,
                float* __restrict__ out, int t)
{
  __shared__ float red[256];
  int b = blockIdx.x, tid = threadIdx.x;
  float acc = 0.f;
  for (int j = tid; j < LSTMU_; j += 256) acc += h1[(size_t)b * LSTMU_ + j] * Ws[j];
  const float* cr = cls_base + (size_t)b * cls_ld;
  for (int v = tid; v < V_; v += 256) acc += cr[v] * Ws[LSTMU_ + v];
  red[tid] = acc; __syncthreads();
  for (int st = 128; st > 0; st >>= 1) { if (tid < st) red[tid] += red[tid + st]; __syncthreads(); }
  if (tid == 0) out[b * T_ + t] = red[0] + bs[0];
}

static inline int nblk_cap(long long total, int thr) {
  long long b = (total + thr - 1) / thr;
  return (int)(b > 32768 ? 32768 : b);
}

extern "C" void kernel_launch(void* const* d_in, const int* in_sizes, int n_in,
                              void* d_out, int out_size, void* d_ws, size_t ws_size,
                              hipStream_t stream)
{
  const int*   tokens = (const int*)  d_in[0];
  const float* memory = (const float*)d_in[1];
  const int*   memlen = (const int*)  d_in[2];
  const float* emb  = (const float*)d_in[3];
  const float* Wp1  = (const float*)d_in[4];
  const float* bp1  = (const float*)d_in[5];
  const float* Wp2  = (const float*)d_in[6];
  const float* bp2  = (const float*)d_in[7];
  const float* Wq   = (const float*)d_in[8];
  const float* Wmem = (const float*)d_in[9];
  const float* ck   = (const float*)d_in[10];
  const float* Wloc = (const float*)d_in[11];
  const float* Wv   = (const float*)d_in[12];
  const float* bv   = (const float*)d_in[13];
  const float* Wa   = (const float*)d_in[14];
  const float* Ua   = (const float*)d_in[15];
  const float* ba   = (const float*)d_in[16];
  const float* W0   = (const float*)d_in[17];
  const float* U0   = (const float*)d_in[18];
  const float* b0   = (const float*)d_in[19];
  const float* W1   = (const float*)d_in[20];
  const float* U1   = (const float*)d_in[21];
  const float* b1   = (const float*)d_in[22];
  const float* Wc   = (const float*)d_in[23];
  const float* bc   = (const float*)d_in[24];
  const float* Ws   = (const float*)d_in[25];
  const float* bs   = (const float*)d_in[26];
  float* out = (float*)d_out;

  // ---- workspace layout (~190 MB) ----
  char* basep = (char*)d_ws;
  size_t off = 0;
  auto alloc = [&](size_t bytes) -> void* {
    void* p = basep + off;
    off = (off + bytes + 255) & ~(size_t)255;
    return p;
  };
  uint16_t* wpk_att = (uint16_t*)alloc((size_t)2304 * 4096 * 2);   // [Wa;Ua] packed
  uint16_t* wpk_0   = (uint16_t*)alloc((size_t)3072 * 4096 * 2);   // [W0;U0]
  uint16_t* wpk_1   = (uint16_t*)alloc((size_t)2048 * 4096 * 2);   // [W1;U1]
  uint16_t* wpk_c   = (uint16_t*)alloc((size_t)2048 * 1008 * 2);   // Wc (N padded to 1008)
  uint16_t* wpk_mem = (uint16_t*)alloc((size_t)1024 * 128  * 2);   // Wmem
  uint16_t* membf   = (uint16_t*)alloc((size_t)B_ * S_ * DENC_ * 2);
  uint16_t* keysbf  = (uint16_t*)alloc((size_t)B_ * S_ * A_ * 2);  // keys as bf16
  float*    states  = (float*)   alloc(((size_t)7 * B_ * LSTMU_ + (size_t)B_ * S_) * 4);
  float* h_a = states + 0 * B_ * LSTMU_;
  float* c_a = states + 1 * B_ * LSTMU_;
  float* h0  = states + 2 * B_ * LSTMU_;
  float* c0  = states + 3 * B_ * LSTMU_;
  float* h1  = states + 4 * B_ * LSTMU_;
  float* c1  = states + 5 * B_ * LSTMU_;
  float* ctx = states + 6 * B_ * LSTMU_;
  float* ast = states + 7 * B_ * LSTMU_;   // [B,S]
  float*    e_buf = (float*)alloc((size_t)B_ * S_ * 4);
  float*    al    = (float*)alloc((size_t)B_ * S_ * 4);
  float*    pq    = (float*)alloc((size_t)B_ * A_ * 4);
  float*    z     = (float*)alloc((size_t)B_ * GATES_ * 4);
  float*    part  = (float*)alloc((size_t)B_ * SCHNK_ * DENC_ * 4); // ctx partials
  uint16_t* xcat  = (uint16_t*)alloc((size_t)B_ * 3072 * 2);        // reused GEMM input

  // ---- one-time setup (re-run every launch; deterministic) ----
  {
    long long n;
    n = (long long)2304 * 4096;
    pack_w<<<nblk_cap(n, 256), 256, 0, stream>>>(Wa, 1280, Ua, 4096, 4096, wpk_att, n);
    n = (long long)3072 * 4096;
    pack_w<<<nblk_cap(n, 256), 256, 0, stream>>>(W0, 2048, U0, 4096, 4096, wpk_0, n);
    n = (long long)2048 * 4096;
    pack_w<<<nblk_cap(n, 256), 256, 0, stream>>>(W1, 1024, U1, 4096, 4096, wpk_1, n);
    n = (long long)2048 * 1008;
    pack_w<<<nblk_cap(n, 256), 256, 0, stream>>>(Wc, 2048, Wc, 1000, 1008, wpk_c, n);
    n = (long long)1024 * 128;
    pack_w<<<nblk_cap(n, 256), 256, 0, stream>>>(Wmem, 1024, Wmem, 128, 128, wpk_mem, n);
    n = (long long)B_ * S_ * DENC_;
    cast_f32_bf16<<<nblk_cap(n, 256), 256, 0, stream>>>(memory, membf, n);
    n = (long long)7 * B_ * LSTMU_ + (long long)B_ * S_;
    zero_f32<<<nblk_cap(n, 256), 256, 0, stream>>>(states, n);
    // keys = memory @ Wmem : [51200,1024] x [1024,128] via WMMA bf16, stored bf16
    wmma_gemm32<<<dim3(A_ / 16, (B_ * S_) / 32), 32, 0, stream>>>(
        membf, DENC_, wpk_mem, A_ / 16, nullptr, (float*)keysbf, (long long)A_, DENC_, A_, 1);
  }

  const int castBlk = (B_ * LSTMU_ + 255) / 256;     // 128 blocks for 32x1024
  const int lstmBlk = (B_ * LSTMU_) / 256;           // 128

  // ---- sequential decoder scan ----
  for (int t = 0; t < T_; ++t) {
    // attention-LSTM input: xcat = [prenet(emb[tok]) | ctx | h_a]  (K = 2304)
    prenet_kernel<<<B_, 256, 0, stream>>>(tokens, t, emb, Wp1, bp1, Wp2, bp2, xcat, 2304);
    cast_rows<<<castBlk, 256, 0, stream>>>(ctx, xcat, 2304, 256, DENC_);
    cast_rows<<<castBlk, 256, 0, stream>>>(h_a, xcat, 2304, 1280, LSTMU_);
    wmma_gemm32<<<dim3(GATES_ / 16, 1), 32, 0, stream>>>(
        xcat, 2304, wpk_att, GATES_ / 16, ba, z, (long long)GATES_, 2304, GATES_, 0);
    lstm_update<<<lstmBlk, 256, 0, stream>>>(z, h_a, c_a);

    // attention
    pq_kernel<<<B_, 128, 0, stream>>>(h_a, Wq, pq);
    energy_kernel<<<(B_ * S_ + 127) / 128, 128, 0, stream>>>(
        keysbf, pq, ast, ck, Wloc, Wv, bv, memlen, e_buf);
    softmax_kernel<<<B_, 256, 0, stream>>>(e_buf, al, ast);
    ctx_partial_kernel<<<dim3(B_, SCHNK_), 128, 0, stream>>>(al, membf, part);
    ctx_reduce_kernel<<<lstmBlk, 256, 0, stream>>>(part, ctx);

    // decoder LSTM 0: xcat = [h_a | ctx | h0]  (K = 3072)
    cast_rows<<<castBlk, 256, 0, stream>>>(h_a, xcat, 3072, 0, LSTMU_);
    cast_rows<<<castBlk, 256, 0, stream>>>(ctx, xcat, 3072, 1024, DENC_);
    cast_rows<<<castBlk, 256, 0, stream>>>(h0,  xcat, 3072, 2048, LSTMU_);
    wmma_gemm32<<<dim3(GATES_ / 16, 1), 32, 0, stream>>>(
        xcat, 3072, wpk_0, GATES_ / 16, b0, z, (long long)GATES_, 3072, GATES_, 0);
    lstm_update<<<lstmBlk, 256, 0, stream>>>(z, h0, c0);

    // decoder LSTM 1: xcat = [h0 | h1]  (K = 2048)
    cast_rows<<<castBlk, 256, 0, stream>>>(h0, xcat, 2048, 0, LSTMU_);
    cast_rows<<<castBlk, 256, 0, stream>>>(h1, xcat, 2048, 1024, LSTMU_);
    wmma_gemm32<<<dim3(GATES_ / 16, 1), 32, 0, stream>>>(
        xcat, 2048, wpk_1, GATES_ / 16, b1, z, (long long)GATES_, 2048, GATES_, 0);
    lstm_update<<<lstmBlk, 256, 0, stream>>>(z, h1, c1);

    // output projection: cls[b,t,:] = [h1 | ctx] @ Wc + bc  (writes straight to d_out)
    cast_rows<<<castBlk, 256, 0, stream>>>(h1,  xcat, 2048, 0, LSTMU_);
    cast_rows<<<castBlk, 256, 0, stream>>>(ctx, xcat, 2048, 1024, DENC_);
    wmma_gemm32<<<dim3(1008 / 16, 1), 32, 0, stream>>>(
        xcat, 2048, wpk_c, 1008 / 16, bc, out + (size_t)t * V_, (long long)T_ * V_, 2048, V_, 0);

    // stop token
    stp_kernel<<<B_, 256, 0, stream>>>(h1, out + (size_t)t * V_, (long long)T_ * V_,
                                       Ws, bs, out + (size_t)B_ * T_ * V_, t);
  }
}
